// CoPEViT_13520557048503
// MI455X (gfx1250) — matrix operand order repaired and checked by reading, not compiled, not run on hardware
//
#include <hip/hip_runtime.h>

typedef __bf16 bf16;
typedef __attribute__((ext_vector_type(16))) __bf16 v16bf;
typedef __attribute__((ext_vector_type(8)))  float  v8f;
typedef unsigned int u32;
typedef __attribute__((ext_vector_type(4))) unsigned int v4u;
typedef __attribute__((ext_vector_type(8))) int v8i_;
typedef __attribute__((ext_vector_type(4))) int v4i_;

#define NPAD 640   // padded N (divisible by 64)
#define NTOK 577   // valid tokens
#define DDIM 768
#define NB   32
#define LDT  40    // LDS tile stride (bf16): 64B data + 16B pad per row

#if defined(__HIP_DEVICE_COMPILE__) && defined(__has_builtin)
# if __has_builtin(__builtin_amdgcn_tensor_load_to_lds) && __has_builtin(__builtin_amdgcn_s_wait_tensorcnt)
#  define USE_TDM 1
# endif
#endif
#ifndef USE_TDM
# define USE_TDM 0
#endif

struct U128x2 { uint4 lo, hi; };

__device__ __forceinline__ v16bf make_frag(uint4 lo, uint4 hi) {
  U128x2 t{lo, hi};
  return __builtin_bit_cast(v16bf, t);
}

// A fragment (16x32, bf16): lane l -> row m0+(l&15); K runs: half*8+[0..7], 16+half*8+[0..7]
__device__ __forceinline__ v16bf frag_a(const bf16* tile, int m0, int lane) {
  int row  = m0 + (lane & 15);
  int half = lane >> 4;
  const bf16* p = tile + row * LDT + half * 8;
  uint4 lo = *(const uint4*)(p);
  uint4 hi = *(const uint4*)(p + 16);
  return make_frag(lo, hi);
}

// B fragment (32x16, bf16), source stored transposed (row n, K contiguous):
// lane l -> col n0+(l&15); K runs: half*8+[0..7], 16+half*8+[0..7]
__device__ __forceinline__ v16bf frag_b(const bf16* tile, int n0, int lane) {
  int row  = n0 + (lane & 15);
  int half = lane >> 4;
  const bf16* p = tile + row * LDT + half * 8;
  uint4 lo = *(const uint4*)(p);
  uint4 hi = *(const uint4*)(p + 16);
  return make_frag(lo, hi);
}

__device__ __forceinline__ v8f wmma_bf16(v16bf a, v16bf b, v8f c) {
  return __builtin_amdgcn_wmma_f32_16x16x32_bf16(false, a, false, b, (short)0, c, false, false);
}

#if USE_TDM
// Issue one TDM 2-D tile load: rows x cols(bf16 elems) from global (row stride stride_elems)
// into LDS at lds_byte_addr, padding each 64B row to 80B (LDT*2 bytes).
__device__ __forceinline__ void tdm_load_tile_2d(u32 lds_byte_addr, const void* gptr,
                                                 int rows, int cols_elems, int stride_elems) {
  unsigned long long ga = (unsigned long long)(size_t)gptr;
  v4u g0 = { 1u,                                   // count=1 (valid user descriptor)
             lds_byte_addr,                        // lds_addr
             (u32)ga,                              // global_addr[31:0]
             ((u32)(ga >> 32) & 0x01FFFFFFu) | (2u << 30) };  // addr[56:32] | type=2
  u32 td0 = (u32)stride_elems;                     // tensor_dim0 (row length)
  u32 td1 = 1u << 20;                              // tensor_dim1: large (no clipping)
  v8i_ g1 = {
    (int)((1u << 16) | (1u << 20) | (3u << 22) | (3u << 25)), // data_size=2B, pad_en, int=16dw, amt=4dw
    (int)((td0 & 0xFFFFu) << 16),                  // atomic_addr=0 | tensor_dim0[15:0]
    (int)((td0 >> 16) | ((td1 & 0xFFFFu) << 16)),  // tensor_dim0[31:16] | tensor_dim1[15:0]
    (int)((td1 >> 16) | ((u32)cols_elems << 16)),  // tensor_dim1[31:16] | tile_dim0
    (int)((u32)rows & 0xFFFFu),                    // tile_dim1 | tile_dim2=0
    (int)((u32)stride_elems),                      // tensor_dim0_stride[31:0]
    0, 0
  };
  v4i_ z4 = {0, 0, 0, 0};
  v8i_ z8 = {0, 0, 0, 0, 0, 0, 0, 0};
  // 6-arg form (clang-23 / therock-10.0 toolchain)
  __builtin_amdgcn_tensor_load_to_lds(g0, g1, z4, z4, z8, 0);
}
__device__ __forceinline__ u32 lds_off(const void* p) {
  return (u32)(unsigned long long)(size_t)p;
}
#endif

// ---------------- Kernel 0: transpose + convert f32 -> bf16 ----------------
__global__ void k_transpose_bf16(const float* __restrict__ src, bf16* __restrict__ dst,
                                 int srcCols, int dstRows) {
  int tid = blockIdx.x * blockDim.x + threadIdx.x;
  int total = dstRows * DDIM;
  if (tid >= total) return;
  int n = tid / DDIM;
  int k = tid % DDIM;
  float v = (n < srcCols) ? src[(long)k * srcCols + n] : 0.0f;
  dst[tid] = (bf16)v;
}

// ---------------- Kernel 1: fused QKV projection ----------------
__global__ __launch_bounds__(256) void k_qkv(
    const float* __restrict__ x,
    const bf16* __restrict__ WtQ, const bf16* __restrict__ WtK, const bf16* __restrict__ WtV,
    const float* __restrict__ bq, const float* __restrict__ bk, const float* __restrict__ bv,
    bf16* __restrict__ Qo, bf16* __restrict__ Ko, bf16* __restrict__ Vt)
{
  __shared__ __align__(16) bf16 As [64 * LDT];
  __shared__ __align__(16) bf16 BsQ[64 * LDT];
  __shared__ __align__(16) bf16 BsK[64 * LDT];
  __shared__ __align__(16) bf16 BsV[64 * LDT];

  const int t = threadIdx.x;
  const int lane = t & 31, wave = t >> 5;
  const int wm = (wave & 3) * 16, wn = (wave >> 2) * 32;
  const int n0 = blockIdx.x * 64;
  const int g0 = blockIdx.y * 64;

  const int lrow = t >> 2;
  const int lk   = (t & 3) * 8;

  int g   = g0 + lrow;
  int b   = g / NPAD, tok = g % NPAD;
  bool av = tok < NTOK;
  const float* xrow = x + (long)(b * NTOK + (av ? tok : 0)) * DDIM;

  v8f aq0 = {}, aq1 = {}, ak0 = {}, ak1 = {}, avv0 = {}, avv1 = {};

  for (int k0 = 0; k0 < DDIM; k0 += 32) {
    union { bf16 h[8]; uint4 v4; } u;
    if (av) {
      float4 p0 = *(const float4*)(xrow + k0 + lk);
      float4 p1 = *(const float4*)(xrow + k0 + lk + 4);
      u.h[0] = (bf16)p0.x; u.h[1] = (bf16)p0.y; u.h[2] = (bf16)p0.z; u.h[3] = (bf16)p0.w;
      u.h[4] = (bf16)p1.x; u.h[5] = (bf16)p1.y; u.h[6] = (bf16)p1.z; u.h[7] = (bf16)p1.w;
      if (k0 + 32 < DDIM) __builtin_prefetch((const void*)(xrow + k0 + 32 + lk), 0, 1);
    } else {
      for (int j = 0; j < 8; ++j) u.h[j] = (bf16)0.0f;
    }
    *(uint4*)&As[lrow * LDT + lk] = u.v4;

    long wofs = (long)(n0 + lrow) * DDIM + k0 + lk;
    *(uint4*)&BsQ[lrow * LDT + lk] = *(const uint4*)(WtQ + wofs);
    *(uint4*)&BsK[lrow * LDT + lk] = *(const uint4*)(WtK + wofs);
    *(uint4*)&BsV[lrow * LDT + lk] = *(const uint4*)(WtV + wofs);

    __syncthreads();

    v16bf a   = frag_a(As, wm, lane);
    v16bf fq0 = frag_b(BsQ, wn,      lane);
    v16bf fq1 = frag_b(BsQ, wn + 16, lane);
    v16bf fk0 = frag_b(BsK, wn,      lane);
    v16bf fk1 = frag_b(BsK, wn + 16, lane);
    v16bf fv0 = frag_b(BsV, wn,      lane);
    v16bf fv1 = frag_b(BsV, wn + 16, lane);

    aq0  = wmma_bf16(a, fq0, aq0);   aq1  = wmma_bf16(a, fq1, aq1);
    ak0  = wmma_bf16(a, fk0, ak0);   ak1  = wmma_bf16(a, fk1, ak1);
    avv0 = wmma_bf16(a, fv0, avv0);  avv1 = wmma_bf16(a, fv1, avv1);

    __syncthreads();
  }

  int colA = lane & 15, half = lane >> 4;
  int nq0 = n0 + wn + colA, nq1 = nq0 + 16;
  float bq0 = bq[nq0], bq1 = bq[nq1];
  float bk0 = bk[nq0], bk1 = bk[nq1];
  float bv0 = bv[nq0], bv1 = bv[nq1];

  for (int r = 0; r < 8; ++r) {
    int grow = g0 + wm + half * 8 + r;
    int bb = grow / NPAD, tk = grow % NPAD;
    bool vld = tk < NTOK;
    long qofs = (long)grow * DDIM + nq0;
    Qo[qofs]      = vld ? (bf16)(aq0[r] + bq0) : (bf16)0.0f;
    Qo[qofs + 16] = vld ? (bf16)(aq1[r] + bq1) : (bf16)0.0f;
    Ko[qofs]      = vld ? (bf16)(ak0[r] + bk0) : (bf16)0.0f;
    Ko[qofs + 16] = vld ? (bf16)(ak1[r] + bk1) : (bf16)0.0f;
    Vt[((long)bb * DDIM + nq0) * NPAD + tk] = vld ? (bf16)(avv0[r] + bv0) : (bf16)0.0f;
    Vt[((long)bb * DDIM + nq1) * NPAD + tk] = vld ? (bf16)(avv1[r] + bv1) : (bf16)0.0f;
  }
}

// ---------------- Kernel 2: fused scores (Q K^T) + logits (Q P) ----------------
__global__ __launch_bounds__(256) void k_scores_logits(
    const bf16* __restrict__ Q, const bf16* __restrict__ K, const bf16* __restrict__ Pt,
    float* __restrict__ scores, float* __restrict__ logits)
{
  __shared__ __align__(16) bf16 As [2][64 * LDT];
  __shared__ __align__(16) bf16 BsK[2][64 * LDT];
  __shared__ __align__(16) bf16 BsP[2][64 * LDT];

  const int t = threadIdx.x, lane = t & 31, wave = t >> 5;
  const int wm = (wave & 3) * 16, wn = (wave >> 2) * 32;
  const int b = blockIdx.z;
  const int q0 = blockIdx.y * 64, m0 = blockIdx.x * 64;

  const bf16* qbase = Q  + ((long)(b * NPAD) + q0) * DDIM;
  const bf16* kbase = K  + ((long)(b * NPAD) + m0) * DDIM;
  const bf16* pbase = Pt + (long)m0 * DDIM;

  v8f s0 = {}, s1 = {}, l0 = {}, l1 = {};
  const int KSTEPS = DDIM / 32;

#if USE_TDM
  // TDM double-buffered staging: wave0->As, wave1->BsK, wave2->BsP
  if (wave == 0) tdm_load_tile_2d(lds_off(&As [0][0]), qbase, 64, 32, DDIM);
  if (wave == 1) tdm_load_tile_2d(lds_off(&BsK[0][0]), kbase, 64, 32, DDIM);
  if (wave == 2) tdm_load_tile_2d(lds_off(&BsP[0][0]), pbase, 64, 32, DDIM);
  if (wave < 3) __builtin_amdgcn_s_wait_tensorcnt(0);
  __syncthreads();

  for (int s = 0; s < KSTEPS; ++s) {
    int cur = s & 1;
    if (s + 1 < KSTEPS) {
      int k1 = (s + 1) * 32;
      if (wave == 0) tdm_load_tile_2d(lds_off(&As [1 - cur][0]), qbase + k1, 64, 32, DDIM);
      if (wave == 1) tdm_load_tile_2d(lds_off(&BsK[1 - cur][0]), kbase + k1, 64, 32, DDIM);
      if (wave == 2) tdm_load_tile_2d(lds_off(&BsP[1 - cur][0]), pbase + k1, 64, 32, DDIM);
    }
    v16bf a   = frag_a(As[cur], wm, lane);
    v16bf k0f = frag_b(BsK[cur], wn, lane);
    v16bf k1f = frag_b(BsK[cur], wn + 16, lane);
    v16bf p0f = frag_b(BsP[cur], wn, lane);
    v16bf p1f = frag_b(BsP[cur], wn + 16, lane);
    s0 = wmma_bf16(a, k0f, s0);  s1 = wmma_bf16(a, k1f, s1);
    l0 = wmma_bf16(a, p0f, l0);  l1 = wmma_bf16(a, p1f, l1);
    if (s + 1 < KSTEPS && wave < 3) __builtin_amdgcn_s_wait_tensorcnt(0);
    __syncthreads();
  }
#else
  const int lrow = t >> 2, lk = (t & 3) * 8;
  for (int s = 0; s < KSTEPS; ++s) {
    int k0 = s * 32;
    *(uint4*)&As [0][lrow * LDT + lk] = *(const uint4*)(qbase + (long)lrow * DDIM + k0 + lk);
    *(uint4*)&BsK[0][lrow * LDT + lk] = *(const uint4*)(kbase + (long)lrow * DDIM + k0 + lk);
    *(uint4*)&BsP[0][lrow * LDT + lk] = *(const uint4*)(pbase + (long)lrow * DDIM + k0 + lk);
    __syncthreads();
    v16bf a   = frag_a(As[0], wm, lane);
    v16bf k0f = frag_b(BsK[0], wn, lane);
    v16bf k1f = frag_b(BsK[0], wn + 16, lane);
    v16bf p0f = frag_b(BsP[0], wn, lane);
    v16bf p1f = frag_b(BsP[0], wn + 16, lane);
    s0 = wmma_bf16(a, k0f, s0);  s1 = wmma_bf16(a, k1f, s1);
    l0 = wmma_bf16(a, p0f, l0);  l1 = wmma_bf16(a, p1f, l1);
    __syncthreads();
  }
#endif

  int colA = lane & 15, half = lane >> 4;
  for (int r = 0; r < 8; ++r) {
    long row = (long)b * NPAD + q0 + wm + half * 8 + r;
    long ofs = row * NPAD + m0 + wn + colA;
    scores[ofs]      = s0[r];  scores[ofs + 16] = s1[r];
    logits[ofs]      = l0[r];  logits[ofs + 16] = l1[r];
  }
}

// ---------------- Kernel 3: CoPE gate/scan/interp + softmax (one block per row) ----------------
__global__ __launch_bounds__(256) void k_cope_softmax(
    const float* __restrict__ scores, const float* __restrict__ logits, bf16* __restrict__ attn)
{
  __shared__ float sh[1024];
  const int q = blockIdx.x, b = blockIdx.y, t = threadIdx.x;
  const long rbase = ((long)b * NPAD + q) * NPAD;
  bf16* arow = attn + rbase;

  if (q >= NTOK) {
    for (int i = t; i < NPAD; i += 256) arow[i] = (bf16)0.0f;
    return;
  }
  const float* srow = scores + rbase;
  const float* lrow = logits + rbase;

  float sc[4] = {0, 0, 0, 0};
  for (int j = 0; j < 4; ++j) {
    int i = t + 256 * j;
    float g = 0.0f;
    if (i < NTOK) {
      sc[j] = srow[i];
      g = (1.0f / (1.0f + __expf(-sc[j]))) * 0.04163054471f;  // sigmoid * 1/sqrt(577)
    }
    sh[i] = g;
  }
  __syncthreads();

  for (int off = 1; off < 1024; off <<= 1) {
    float v[4];
    for (int j = 0; j < 4; ++j) { int i = t + 256 * j; v[j] = (i >= off) ? sh[i - off] : 0.0f; }
    __syncthreads();
    for (int j = 0; j < 4; ++j) { int i = t + 256 * j; sh[i] += v[j]; }
    __syncthreads();
  }

  float T = sh[NTOK - 1];
  float sex[4] = {0, 0, 0, 0};
  for (int j = 0; j < 4; ++j) { int i = t + 256 * j; if (i < NTOK) sex[j] = (i > 0) ? sh[i - 1] : 0.0f; }
  __syncthreads();

  float vals[4] = {0, 0, 0, 0};
  float vmax = -3.0e38f;
  for (int j = 0; j < 4; ++j) {
    int i = t + 256 * j;
    if (i < NTOK) {
      float pos = fminf(T - sex[j], (float)(NTOK - 1));
      float pc = ceilf(pos), pf = floorf(pos);
      int ic  = (int)pc; if (ic < 0) ic = 0; if (ic > NTOK - 1) ic = NTOK - 1;
      int ifl = (int)pf; if (ifl < 0) ifl = 0; if (ifl > NTOK - 1) ifl = NTOK - 1;
      float lc = lrow[ic], lf = lrow[ifl];
      float w = pc - pos;
      vals[j] = sc[j] + w * lc + (1.0f - w) * lf;
      vmax = fmaxf(vmax, vals[j]);
    }
  }

  sh[t] = vmax; __syncthreads();
  for (int off = 128; off > 0; off >>= 1) { if (t < off) sh[t] = fmaxf(sh[t], sh[t + off]); __syncthreads(); }
  float M = sh[0]; __syncthreads();

  float ex[4] = {0, 0, 0, 0};
  float ssum = 0.0f;
  for (int j = 0; j < 4; ++j) { int i = t + 256 * j; if (i < NTOK) { ex[j] = __expf(vals[j] - M); ssum += ex[j]; } }
  sh[t] = ssum; __syncthreads();
  for (int off = 128; off > 0; off >>= 1) { if (t < off) sh[t] += sh[t + off]; __syncthreads(); }
  float inv = 1.0f / sh[0];

  for (int j = 0; j < 4; ++j) {
    int i = t + 256 * j;
    if (i < NPAD) arow[i] = (i < NTOK) ? (bf16)(ex[j] * inv) : (bf16)0.0f;
  }
}

// ---------------- Kernel 4: out = attn @ V ----------------
__global__ __launch_bounds__(256) void k_av(
    const bf16* __restrict__ attn, const bf16* __restrict__ Vt, float* __restrict__ out)
{
  __shared__ __align__(16) bf16 As[2][64 * LDT];
  __shared__ __align__(16) bf16 Bs[2][64 * LDT];

  const int t = threadIdx.x, lane = t & 31, wave = t >> 5;
  const int wm = (wave & 3) * 16, wn = (wave >> 2) * 32;
  const int b = blockIdx.z, q0 = blockIdx.y * 64, d0 = blockIdx.x * 64;

  const bf16* abase = attn + ((long)(b * NPAD) + q0) * NPAD;
  const bf16* vbase = Vt   + ((long)b * DDIM + d0) * NPAD;

  v8f c0 = {}, c1 = {};
  const int KSTEPS = NPAD / 32;

#if USE_TDM
  if (wave == 0) tdm_load_tile_2d(lds_off(&As[0][0]), abase, 64, 32, NPAD);
  if (wave == 1) tdm_load_tile_2d(lds_off(&Bs[0][0]), vbase, 64, 32, NPAD);
  if (wave < 2) __builtin_amdgcn_s_wait_tensorcnt(0);
  __syncthreads();

  for (int s = 0; s < KSTEPS; ++s) {
    int cur = s & 1;
    if (s + 1 < KSTEPS) {
      int k1 = (s + 1) * 32;
      if (wave == 0) tdm_load_tile_2d(lds_off(&As[1 - cur][0]), abase + k1, 64, 32, NPAD);
      if (wave == 1) tdm_load_tile_2d(lds_off(&Bs[1 - cur][0]), vbase + k1, 64, 32, NPAD);
    }
    v16bf a  = frag_a(As[cur], wm, lane);
    v16bf b0 = frag_b(Bs[cur], wn, lane);
    v16bf b1 = frag_b(Bs[cur], wn + 16, lane);
    c0 = wmma_bf16(a, b0, c0);
    c1 = wmma_bf16(a, b1, c1);
    if (s + 1 < KSTEPS && wave < 2) __builtin_amdgcn_s_wait_tensorcnt(0);
    __syncthreads();
  }
#else
  const int lrow = t >> 2, lk = (t & 3) * 8;
  for (int s = 0; s < KSTEPS; ++s) {
    int k0 = s * 32;
    *(uint4*)&As[0][lrow * LDT + lk] = *(const uint4*)(abase + (long)lrow * NPAD + k0 + lk);
    *(uint4*)&Bs[0][lrow * LDT + lk] = *(const uint4*)(vbase + (long)lrow * NPAD + k0 + lk);
    __syncthreads();
    v16bf a  = frag_a(As[0], wm, lane);
    v16bf b0 = frag_b(Bs[0], wn, lane);
    v16bf b1 = frag_b(Bs[0], wn + 16, lane);
    c0 = wmma_bf16(a, b0, c0);
    c1 = wmma_bf16(a, b1, c1);
    __syncthreads();
  }
#endif

  int colA = lane & 15, half = lane >> 4;
  for (int r = 0; r < 8; ++r) {
    int q = q0 + wm + half * 8 + r;
    if (q < NTOK) {
      long ofs = ((long)(b * NTOK) + q) * DDIM + d0 + wn + colA;
      out[ofs]      = c0[r];
      out[ofs + 16] = c1[r];
    }
  }
}

// ---------------- Host launch ----------------
extern "C" void kernel_launch(void* const* d_in, const int* in_sizes, int n_in,
                              void* d_out, int out_size, void* d_ws, size_t ws_size,
                              hipStream_t stream) {
  (void)in_sizes; (void)n_in; (void)out_size; (void)ws_size;
  const float* x  = (const float*)d_in[0];
  const float* Wq = (const float*)d_in[1];
  const float* bq = (const float*)d_in[2];
  const float* Wk = (const float*)d_in[3];
  const float* bk = (const float*)d_in[4];
  const float* Wv = (const float*)d_in[5];
  const float* bv = (const float*)d_in[6];
  const float* pe = (const float*)d_in[7];
  float* out = (float*)d_out;

  char* ws = (char*)d_ws;
  size_t ofs = 0;
  auto alloc = [&](size_t bytes) -> char* {
    char* p = ws + ofs;
    ofs += (bytes + 255) & ~((size_t)255);
    return p;
  };

  bf16* WtQ = (bf16*)alloc((size_t)DDIM * DDIM * 2);
  bf16* WtK = (bf16*)alloc((size_t)DDIM * DDIM * 2);
  bf16* WtV = (bf16*)alloc((size_t)DDIM * DDIM * 2);
  bf16* Pt  = (bf16*)alloc((size_t)NPAD * DDIM * 2);
  bf16* Qb  = (bf16*)alloc((size_t)NB * NPAD * DDIM * 2);
  bf16* Kb  = (bf16*)alloc((size_t)NB * NPAD * DDIM * 2);
  bf16* Vt  = (bf16*)alloc((size_t)NB * DDIM * NPAD * 2);
  float* scores = (float*)alloc((size_t)NB * NPAD * NPAD * 4);
  float* logits = (float*)alloc((size_t)NB * NPAD * NPAD * 4);
  bf16* attn = (bf16*)alloc((size_t)NB * NPAD * NPAD * 2);

  int wtot = DDIM * DDIM;
  k_transpose_bf16<<<(wtot + 255) / 256, 256, 0, stream>>>(Wq, WtQ, DDIM, DDIM);
  k_transpose_bf16<<<(wtot + 255) / 256, 256, 0, stream>>>(Wk, WtK, DDIM, DDIM);
  k_transpose_bf16<<<(wtot + 255) / 256, 256, 0, stream>>>(Wv, WtV, DDIM, DDIM);
  int ptot = NPAD * DDIM;
  k_transpose_bf16<<<(ptot + 255) / 256, 256, 0, stream>>>(pe, Pt, NTOK, NPAD);

  k_qkv<<<dim3(DDIM / 64, (NB * NPAD) / 64, 1), 256, 0, stream>>>(
      x, WtQ, WtK, WtV, bq, bk, bv, Qb, Kb, Vt);

  k_scores_logits<<<dim3(NPAD / 64, NPAD / 64, NB), 256, 0, stream>>>(
      Qb, Kb, Pt, scores, logits);

  k_cope_softmax<<<dim3(NPAD, NB, 1), 256, 0, stream>>>(scores, logits, attn);

  k_av<<<dim3(DDIM / 64, NPAD / 64, NB), 256, 0, stream>>>(attn, Vt, out);
}
